// GSNN_Decoder_11106785427521
// MI455X (gfx1250) — matrix operand order repaired and checked by reading, not compiled, not run on hardware
//
#include <hip/hip_runtime.h>
#include <hip/hip_bf16.h>

// ---------------------------------------------------------------------------
// GSNN decoder for MI455X (gfx1250, wave32, WMMA, WGP LDS).
//   h  = relu(x @ W1 + b1); hz = [h | z]; hn = hz / (||hz|| + 1e-6)   (bf16)
//   a1 = spmm(hn)            (f32 atomics)
//   g2 = relu(a1 @ W2 + b2)  (bf16)
//   a2 = spmm(g2)            (f32 atomics)
//   y  = a2 @ W3 + b3        (f32 out)
// GEMMs: v_wmma_f32_16x16x32_bf16. B fragments staged block-wide through
// double-buffered LDS (4KB/K-step, shared by 8 waves); A fragments
// double-buffered in registers with in-flight f32->bf16 conversion.
// SpMM gathers a bf16 matrix (76.8MB) resident in the 192MB L2.
// ---------------------------------------------------------------------------

typedef __attribute__((ext_vector_type(16))) __bf16 v16bf;
typedef __attribute__((ext_vector_type(8)))  float  v8f;

#define NNODES   50000
#define NEDGES   1600000
#define XDIM     512
#define HDIM     512
#define ZDIM     256
#define HZDIM    768
#define YDIM     256

// ---------------------------------------------------------------------------
// Pack row-major f32 weight [K, M] into per-lane WMMA B fragments (bf16).
// Fragment id f = (kt*ntiles + nt); element layout matches ISA 7.12.2:
//   lane<16 : col = nt*16+lane,    K = kt*32 + (e<8 ? e : 8 + e)      (e = 0..15)
//   lane>=16: col = nt*16+lane-16, K = kt*32 + 8 + (e<8 ? e : 8 + e)
// Stored as Bp[(f*32 + lane)*16 + e].
// ---------------------------------------------------------------------------
__global__ __launch_bounds__(256) void pack_w_kernel(
    const float* __restrict__ W, __bf16* __restrict__ Bp, int K, int M)
{
    const int ntiles = M >> 4;
    int tid = blockIdx.x * 256 + threadIdx.x;
    if (tid >= K * M) return;
    int e    = tid & 15;
    int lane = (tid >> 4) & 31;
    int f    = tid >> 9;
    int nt   = f % ntiles;
    int kt   = f / ntiles;
    int k    = kt * 32 + ((lane >> 4) << 3) + (e & 7) + ((e >> 3) << 4);
    int col  = nt * 16 + (lane & 15);
    Bp[tid] = (__bf16)W[(size_t)k * M + col];
}

// ---------------------------------------------------------------------------
// WMMA GEMM: D = act(A[Mrows,K] @ Bp + bias).  A: row-major f32.
// Block: 8 waves -> 128 rows x 64 cols. Wave: 16 rows x 64 cols (4 accs).
// B staged through double-buffered LDS (ds_load_b128 feed to the WMMAs);
// A double-buffered in registers. KSTEPS compile-time -> full unroll.
// ---------------------------------------------------------------------------
template <int KSTEPS>
__global__ __launch_bounds__(256) void gemm_bf16_kernel(
    const float* __restrict__ A, const __bf16* __restrict__ Bp,
    const float* __restrict__ bias,
    float* __restrict__ Df32, __bf16* __restrict__ Dbf16,
    int Mrows, int Ncols, int relu)
{
    const int tid  = threadIdx.x;
    const int lane = tid & 31;
    const int wave = tid >> 5;
    const int half = lane >> 4;     // 0 or 1
    const int mr   = lane & 15;

    const int numRowTiles = Mrows >> 4;
    const int rowTileRaw  = blockIdx.x * 8 + wave;
    const bool valid      = rowTileRaw < numRowTiles;
    // Clamp instead of early-return: every wave stays alive for the
    // block-wide LDS staging barriers (redundant compute, guarded stores).
    const int rowTile     = valid ? rowTileRaw : (numRowTiles - 1);

    const int ntiles = Ncols >> 4;
    const int ntBase = blockIdx.y * 4;            // 4 col-tiles of 16 = 64 cols
    const int K = KSTEPS * 32;

    const float* __restrict__ arow = A + (size_t)(rowTile * 16 + mr) * K;
    const v16bf* __restrict__ bfrag0 = (const v16bf*)Bp;

    // Double-buffered LDS for the block's B fragments: 2 x 4KB.
    __shared__ uint4 ldsB[2][256];

    // A fragment: 16 bf16/lane from two 32B row chunks (ISA A layout),
    // converted f32->bf16 in-register (v_cvt_pk_bf16_f32).
    auto loadA = [&](int kt, v16bf& af) {
        const int k0 = kt * 32 + half * 8;
        const float4* p0 = (const float4*)(arow + k0);
        const float4* p1 = (const float4*)(arow + k0 + 16);
        float4 f0 = p0[0], f1 = p0[1];
        float4 f2 = p1[0], f3 = p1[1];
        af[0]  = (__bf16)f0.x; af[1]  = (__bf16)f0.y;
        af[2]  = (__bf16)f0.z; af[3]  = (__bf16)f0.w;
        af[4]  = (__bf16)f1.x; af[5]  = (__bf16)f1.y;
        af[6]  = (__bf16)f1.z; af[7]  = (__bf16)f1.w;
        af[8]  = (__bf16)f2.x; af[9]  = (__bf16)f2.y;
        af[10] = (__bf16)f2.z; af[11] = (__bf16)f2.w;
        af[12] = (__bf16)f3.x; af[13] = (__bf16)f3.y;
        af[14] = (__bf16)f3.z; af[15] = (__bf16)f3.w;
    };
    // Block-cooperative B staging: 256 threads x 16B = 4KB per K-step.
    auto stageLoad = [&](int kt) -> uint4 {
        const uint4* src = (const uint4*)(bfrag0 + (size_t)(kt * ntiles + ntBase) * 32);
        return src[tid];
    };

    v8f acc[4] = {};
    v16bf aF[2];
    uint4 stg;

    // Preamble: buffer 0 staged, A(0) in regs, global load for K-step 1 in flight.
    stg = stageLoad(0);
    loadA(0, aF[0]);
    ldsB[0][tid] = stg;
    if (KSTEPS > 1) stg = stageLoad(1);

#pragma unroll
    for (int kt = 0; kt < KSTEPS; ++kt) {
        const int cur = kt & 1;
        const int nxt = cur ^ 1;

        __syncthreads();                           // ldsB[cur] ready for all waves

        if (kt + 1 < KSTEPS) loadA(kt + 1, aF[nxt]);

        const v16bf* lb = (const v16bf*)&ldsB[cur][0];
#pragma unroll
        for (int c = 0; c < 4; ++c) {
            v16bf bf = lb[c * 32 + lane];          // ds_load_b128 x2, low latency
            acc[c] = __builtin_amdgcn_wmma_f32_16x16x32_bf16(
                false, aF[cur], false, bf, (short)0, acc[c], false, false);
        }

        // After the WMMAs: commit the prefetched B to the other buffer and
        // issue the next global staging load (vmem wait never blocks a WMMA).
        if (kt + 1 < KSTEPS) {
            ldsB[nxt][tid] = stg;
            if (kt + 2 < KSTEPS) stg = stageLoad(kt + 2);
        }
    }

    if (!valid) return;

    // Epilogue: C/D layout — VGPR i: lanes0-15 M=i, lanes16-31 M=8+i; N=lane&15
#pragma unroll
    for (int c = 0; c < 4; ++c) {
        const int col = (ntBase + c) * 16 + mr;
        const float bv = bias ? bias[col] : 0.0f;
#pragma unroll
        for (int i = 0; i < 8; ++i) {
            const int row = rowTile * 16 + half * 8 + i;
            float v = acc[c][i] + bv;
            if (relu) v = fmaxf(v, 0.0f);
            if (Df32)  Df32 [(size_t)row * Ncols + col] = v;
            if (Dbf16) Dbf16[(size_t)row * Ncols + col] = (__bf16)v;
        }
    }
}

// ---------------------------------------------------------------------------
// Row-normalize [h | z] and emit bf16 gather matrix hn[N, 768].
// One 256-thread block per row; z's norm contribution folded into reduction.
// ---------------------------------------------------------------------------
__global__ __launch_bounds__(256) void norm_concat_kernel(
    const float* __restrict__ t1, const float* __restrict__ z,
    __bf16* __restrict__ hn)
{
    const int row = blockIdx.x;
    const int t   = threadIdx.x;
    __shared__ float red[256];

    const float* __restrict__ hr = t1 + (size_t)row * HDIM;
    const float a  = hr[t];
    const float b  = hr[t + 256];
    const float zz = z[t];
    red[t] = a * a + b * b + zz * zz;
    __syncthreads();
#pragma unroll
    for (int s = 128; s > 0; s >>= 1) {
        if (t < s) red[t] += red[t + s];
        __syncthreads();
    }
    const float inv = 1.0f / (sqrtf(red[0]) + 1e-6f);

    __bf16* __restrict__ orow = hn + (size_t)row * HZDIM;
    orow[t]       = (__bf16)(a * inv);
    orow[t + 256] = (__bf16)(b * inv);
    orow[t + 512] = (__bf16)(zz * inv);
}

// ---------------------------------------------------------------------------
// SpMM: out[rows[e]] += vals[e] * H[cols[e]]  (H bf16, L2-resident; f32 atomics)
// One wave per edge, 8 edges per block. bf16 pairs loaded as packed dwords
// (12 b32 loads/lane instead of 24 u16 loads), expanded exactly via shifts.
// ---------------------------------------------------------------------------
__global__ __launch_bounds__(256) void spmm_atomic_kernel(
    const int* __restrict__ rows, const int* __restrict__ cols,
    const float* __restrict__ vals, const __bf16* __restrict__ H,
    float* __restrict__ out, int nE)
{
    const int lane = threadIdx.x & 31;
    const int wave = threadIdx.x >> 5;
    const int e = blockIdx.x * 8 + wave;
    if (e >= nE) return;
    const int r = rows[e];
    const int c = cols[e];
    const float v = vals[e];
    const unsigned int* __restrict__ h32 =
        (const unsigned int*)(H + (size_t)c * HZDIM);
    float* __restrict__ o = out + (size_t)r * HZDIM;
#pragma unroll
    for (int k = 0; k < HZDIM / 64; ++k) {        // 12 packed dwords per lane
        const int p = k * 32 + lane;              // dword index into row
        const unsigned int w = h32[p];
        const float lo = __uint_as_float(w << 16);           // bf16 -> f32 exact
        const float hi = __uint_as_float(w & 0xffff0000u);
        atomicAdd(&o[2 * p],     v * lo);
        atomicAdd(&o[2 * p + 1], v * hi);
    }
}

__global__ __launch_bounds__(256) void zero_f32_kernel(float* __restrict__ p, size_t n)
{
    size_t i = (size_t)blockIdx.x * 256 + threadIdx.x;
    const size_t stride = (size_t)gridDim.x * 256;
    for (; i < n; i += stride) p[i] = 0.0f;
}

// ---------------------------------------------------------------------------
extern "C" void kernel_launch(void* const* d_in, const int* in_sizes, int n_in,
                              void* d_out, int out_size, void* d_ws, size_t ws_size,
                              hipStream_t stream)
{
    (void)in_sizes; (void)n_in; (void)out_size; (void)ws_size;

    const float* x  = (const float*)d_in[0];
    const float* z  = (const float*)d_in[1];
    const int*   er = (const int*)  d_in[2];
    const int*   ec = (const int*)  d_in[3];
    const float* ev = (const float*)d_in[4];
    const float* W1 = (const float*)d_in[5];
    const float* b1 = (const float*)d_in[6];
    const float* W2 = (const float*)d_in[7];
    const float* b2 = (const float*)d_in[8];
    const float* W3 = (const float*)d_in[9];
    const float* b3 = (const float*)d_in[10];
    float* y = (float*)d_out;

    // Workspace layout (all offsets 256B-aligned):
    char* ws = (char*)d_ws;
    const size_t szACC = (size_t)NNODES * HZDIM * 4;   // 153.6 MB (a1, then a2)
    const size_t szT1  = (size_t)NNODES * HDIM  * 4;   // 102.4 MB (t1, then g2)
    const size_t szHN  = (size_t)NNODES * HZDIM * 2;   //  76.8 MB
    float*  accF = (float*) (ws);
    float*  t1   = (float*) (ws + szACC);
    __bf16* g2   = (__bf16*)(ws + szACC);              // reuses t1 region after t1 dies
    __bf16* hn   = (__bf16*)(ws + szACC + szT1);
    __bf16* W1p  = (__bf16*)(ws + szACC + szT1 + szHN);
    __bf16* W2p  = (__bf16*)((char*)W1p + (size_t)XDIM  * HDIM  * 2);
    __bf16* W3p  = (__bf16*)((char*)W2p + (size_t)HZDIM * HZDIM * 2);

    const int rowBlocks = (NNODES / 16 + 7) / 8;       // 3125 row tiles / 8 waves

    // Pack weights into WMMA B-fragment layout (bf16).
    pack_w_kernel<<<(XDIM  * HDIM  + 255) / 256, 256, 0, stream>>>(W1, W1p, XDIM,  HDIM);
    pack_w_kernel<<<(HZDIM * HZDIM + 255) / 256, 256, 0, stream>>>(W2, W2p, HZDIM, HZDIM);
    pack_w_kernel<<<(HZDIM * YDIM  + 255) / 256, 256, 0, stream>>>(W3, W3p, HZDIM, YDIM);

    // Stage 1: t1 = relu(x @ W1 + b1)   [N, 512] f32   (K = 512 -> 16 K-steps)
    gemm_bf16_kernel<16><<<dim3(rowBlocks, HDIM / 64), 256, 0, stream>>>(
        x, W1p, b1, t1, nullptr, NNODES, HDIM, 1);

    // Stage 2: hn = normalize([t1 | z]) -> bf16 [N, 768]
    norm_concat_kernel<<<NNODES, 256, 0, stream>>>(t1, z, hn);

    // Stage 3: a1 = spmm(hn)
    zero_f32_kernel<<<8192, 256, 0, stream>>>(accF, (size_t)NNODES * HZDIM);
    spmm_atomic_kernel<<<(NEDGES + 7) / 8, 256, 0, stream>>>(er, ec, ev, hn, accF, NEDGES);

    // Stage 4: g2 = relu(a1 @ W2 + b2) -> bf16 [N, 768]   (K = 768 -> 24 K-steps)
    gemm_bf16_kernel<24><<<dim3(rowBlocks, HZDIM / 64), 256, 0, stream>>>(
        accF, W2p, b2, nullptr, g2, NNODES, HZDIM, 1);

    // Stage 5: a2 = spmm(g2)   (accumulator reused; stream order guarantees safety)
    zero_f32_kernel<<<8192, 256, 0, stream>>>(accF, (size_t)NNODES * HZDIM);
    spmm_atomic_kernel<<<(NEDGES + 7) / 8, 256, 0, stream>>>(er, ec, ev, g2, accF, NEDGES);

    // Stage 6: y = a2 @ W3 + b3   [N, 256] f32   (K = 768 -> 24 K-steps)
    gemm_bf16_kernel<24><<<dim3(rowBlocks, YDIM / 64), 256, 0, stream>>>(
        accF, W3p, b3, y, nullptr, NNODES, YDIM, 0);
}
